// Attention_31104153157891
// MI455X (gfx1250) — matrix operand order repaired and checked by reading, not compiled, not run on hardware
//
#include <hip/hip_runtime.h>
#include <hip/hip_bf16.h>
#include <math.h>

// ---------------------------------------------------------------------------
// Types for CDNA5 WMMA
// ---------------------------------------------------------------------------
typedef __attribute__((ext_vector_type(16))) __bf16 v16bf;
typedef __attribute__((ext_vector_type(8)))  float  v8f;

union ABOp {
    v16bf v;
    uint4 q[2];
};

// round-to-nearest-even f32 -> bf16
__device__ __forceinline__ unsigned short f2bf(float f) {
    unsigned u = __builtin_bit_cast(unsigned, f);
    unsigned rounding = 0x7FFFu + ((u >> 16) & 1u);
    return (unsigned short)((u + rounding) >> 16);
}
__device__ __forceinline__ unsigned pack_bf16(float lo, float hi) {
    return (unsigned)f2bf(lo) | ((unsigned)f2bf(hi) << 16);
}

// Hardware tanh (gfx1250 V_TANH_F32 trans op). Builtin if available,
// else inline asm with a trailing independent op for the trans hazard.
__device__ __forceinline__ float fast_tanh(float x) {
#if __has_builtin(__builtin_amdgcn_tanhf)
    return __builtin_amdgcn_tanhf(x);
#else
    float r;
    asm volatile("v_tanh_f32 %0, %1\n\tv_nop" : "=v"(r) : "v"(x));
    return r;
#endif
}

// Problem constants (B=32, S=4096, H=512, L=2)
#define BATCH 32
#define SEQ   4096
#define HDIM  512
#define NROWS (BATCH * SEQ)          // 131072 flattened (b,s) rows
#define LDS_ROW_BYTES 1040           // 512 bf16 = 1024B + 16B pad (bank spread)

// ---------------------------------------------------------------------------
// Kernel 0a: convert W (HxH f32, row-major over output channel o) to bf16
// ---------------------------------------------------------------------------
__global__ void k0_convert_w(const float* __restrict__ W,
                             unsigned short* __restrict__ Wb) {
    int i = blockIdx.x * 256 + threadIdx.x;   // grid covers 512*512
    Wb[i] = f2bf(W[i]);
}

// ---------------------------------------------------------------------------
// Kernel 0b: hb[b][n] = bias[n] + hidden[last=1][b][n]   (32 x 512 table)
// ---------------------------------------------------------------------------
__global__ void k0_bias(const float* __restrict__ hidden,
                        const float* __restrict__ bias,
                        float* __restrict__ hb) {
    const int b = blockIdx.x;
    const int n0 = threadIdx.x * 2;
    float2 bv = *(const float2*)(bias + n0);
    float2 hv = *(const float2*)(hidden + (size_t)(BATCH + b) * HDIM + n0);
    float2 r;
    r.x = bv.x + hv.x;
    r.y = bv.y + hv.y;
    *(float2*)(hb + (size_t)b * HDIM + n0) = r;
}

// ---------------------------------------------------------------------------
// Kernel 1: scores[b,s] = sum_o tanh( (out @ W^T)[b,s,o] + hb[b,o] ) * v[o]
// 64 threads (2 waves) per block; block handles 32 rows; wave handles 16 rows.
// ---------------------------------------------------------------------------
__global__ __launch_bounds__(64) void k1_scores(
    const float* __restrict__ out,            // (B,S,H) f32
    const unsigned short* __restrict__ Wb,    // (H,H) bf16
    const float* __restrict__ hb,             // (B,H) bias + h_last
    const float* __restrict__ vvec,           // (H)
    float* __restrict__ scores)               // (B*S)
{
    __shared__ __align__(16) unsigned char lds[32 * LDS_ROW_BYTES];

    const int tid     = threadIdx.x;
    const int rowbase = blockIdx.x * 32;      // 32 rows per block, never crosses b
    const int b_idx   = rowbase >> 12;        // row / 4096

    // ---- Phase 1: stage 32x512 f32 tile of `out` into LDS as bf16 ----
    for (int i = tid; i < 32 * 128; i += 64) {          // 128 float4 per row
        const int r  = i >> 7;
        const int c4 = i & 127;
        const float4 f = *(const float4*)(out + (size_t)(rowbase + r) * HDIM + c4 * 4);
        uint2 p;
        p.x = pack_bf16(f.x, f.y);
        p.y = pack_bf16(f.z, f.w);
        *(uint2*)&lds[r * LDS_ROW_BYTES + c4 * 8] = p;
    }
    __syncthreads();

    const int lane = tid & 31;
    const int wv   = tid >> 5;                // wave in block (0,1)
    const int nlo  = lane & 15;
    const int hi   = lane >> 4;               // half-wave select
    const int rows16 = rowbase + wv * 16;

    // A-operand LDS base for this lane (row = wv*16 + nlo; hi lanes K+8 => +16B)
    const unsigned char* abase = &lds[(wv * 16 + nlo) * LDS_ROW_BYTES + hi * 16];
    const unsigned char* wbytes = (const unsigned char*)Wb;
    const float* hbrow = hb + (size_t)b_idx * HDIM;

    float sacc[8] = {0.f, 0.f, 0.f, 0.f, 0.f, 0.f, 0.f, 0.f};

    for (int nc = 0; nc < 4; ++nc) {                    // 4 x 128 output cols
        v8f acc[8] = {};                                // 8 N-tiles of 16 cols

        for (int ks = 0; ks < 16; ++ks) {               // K = 16 x 32
            ABOp a;
            a.q[0] = *(const uint4*)(abase + ks * 64);       // K 0-7  (+8 hi)
            a.q[1] = *(const uint4*)(abase + ks * 64 + 32);  // K 16-23(+8 hi)

#pragma unroll
            for (int t = 0; t < 8; ++t) {
                const int n = nc * 128 + t * 16 + nlo;       // output col per lane
                const unsigned char* bp =
                    wbytes + (size_t)n * (HDIM * 2) + ks * 64 + hi * 16;
                ABOp bm;
                bm.q[0] = *(const uint4*)(bp);               // K 0-7  (+8 hi)
                bm.q[1] = *(const uint4*)(bp + 32);          // K 16-23(+8 hi)
                acc[t] = __builtin_amdgcn_wmma_f32_16x16x32_bf16(
                    false, a.v, false, bm.v, (short)0, acc[t], false, false);
            }
        }

        // hw-tanh + dot with v; lane holds col n, rows r (+8 for hi lanes)
#pragma unroll
        for (int t = 0; t < 8; ++t) {
            const int n   = nc * 128 + t * 16 + nlo;
            const float hbn = hbrow[n];
            const float vn  = vvec[n];
#pragma unroll
            for (int r = 0; r < 8; ++r) {
                sacc[r] += vn * fast_tanh(acc[t][r] + hbn);
            }
        }
    }

    // reduce across the 16 lanes of each half-wave (cols 0..15)
#pragma unroll
    for (int r = 0; r < 8; ++r) {
        float x = sacc[r];
        x += __shfl_xor(x, 1, 32);
        x += __shfl_xor(x, 2, 32);
        x += __shfl_xor(x, 4, 32);
        x += __shfl_xor(x, 8, 32);
        sacc[r] = x;
    }

    if (nlo == 0) {  // lane 0 -> rows 0-7, lane 16 -> rows 8-15
        float* sp = scores + rows16 + hi * 8;
        *(float4*)(sp)     = make_float4(sacc[0], sacc[1], sacc[2], sacc[3]);
        *(float4*)(sp + 4) = make_float4(sacc[4], sacc[5], sacc[6], sacc[7]);
    }
}

// ---------------------------------------------------------------------------
// Kernel 2: row softmax over S=4096 per batch
// ---------------------------------------------------------------------------
__global__ __launch_bounds__(256) void k2_softmax(const float* __restrict__ scores,
                                                  float* __restrict__ attn) {
    __shared__ float red[256];
    const int b   = blockIdx.x;
    const int tid = threadIdx.x;
    const float* s = scores + (size_t)b * SEQ;

    float m = -INFINITY;
    for (int i = tid; i < SEQ; i += 256) m = fmaxf(m, s[i]);
    red[tid] = m;
    __syncthreads();
    for (int off = 128; off > 0; off >>= 1) {
        if (tid < off) red[tid] = fmaxf(red[tid], red[tid + off]);
        __syncthreads();
    }
    m = red[0];
    __syncthreads();

    float sum = 0.f;
    for (int i = tid; i < SEQ; i += 256) sum += __expf(s[i] - m);
    red[tid] = sum;
    __syncthreads();
    for (int off = 128; off > 0; off >>= 1) {
        if (tid < off) red[tid] += red[tid + off];
        __syncthreads();
    }
    const float inv = 1.0f / red[0];

    float* a = attn + (size_t)b * SEQ;
    for (int i = tid; i < SEQ; i += 256) a[i] = __expf(s[i] - m) * inv;
}

// ---------------------------------------------------------------------------
// Kernel 3: context[b,h] = sum_s attn[b,s] * out[b,s,h]
// grid (H/128, B); 256 threads: 128 h-cols x 2 s-partitions (coalesced over h)
// ---------------------------------------------------------------------------
__global__ __launch_bounds__(256) void k3_context(const float* __restrict__ out,
                                                  const float* __restrict__ attn,
                                                  float* __restrict__ ctx) {
    __shared__ float sm[256];
    const int b    = blockIdx.y;
    const int h    = blockIdx.x * 128 + (threadIdx.x & 127);
    const int part = threadIdx.x >> 7;

    const float* op = out + (size_t)b * SEQ * HDIM + h;
    const float* ap = attn + (size_t)b * SEQ;

    float sum = 0.f;
    for (int s = part; s < SEQ; s += 2)
        sum += ap[s] * op[(size_t)s * HDIM];

    sm[threadIdx.x] = sum;
    __syncthreads();
    if (part == 0)
        ctx[b * HDIM + h] = sm[threadIdx.x] + sm[threadIdx.x + 128];
}

// ---------------------------------------------------------------------------
// Launch
// ---------------------------------------------------------------------------
extern "C" void kernel_launch(void* const* d_in, const int* in_sizes, int n_in,
                              void* d_out, int out_size, void* d_ws, size_t ws_size,
                              hipStream_t stream) {
    const float* out_t  = (const float*)d_in[0];   // (32,4096,512)
    const float* hidden = (const float*)d_in[1];   // (2,32,512)
    const float* W      = (const float*)d_in[2];   // (512,512)
    const float* bias   = (const float*)d_in[3];   // (512)
    const float* vvec   = (const float*)d_in[4];   // (512)

    float* ctx  = (float*)d_out;                   // 32*512
    float* attn = (float*)d_out + BATCH * HDIM;    // 32*4096

    // workspace layout: W bf16 [0,512K) | scores [512K,1M) | hb [1M,1M+64K)
    unsigned short* Wb = (unsigned short*)d_ws;
    float* scores = (float*)((char*)d_ws + (size_t)HDIM * HDIM * 2);
    float* hb     = (float*)((char*)d_ws + (size_t)HDIM * HDIM * 2
                                         + (size_t)NROWS * 4);

    k0_convert_w<<<(HDIM * HDIM) / 256, 256, 0, stream>>>(W, Wb);
    k0_bias<<<BATCH, 256, 0, stream>>>(hidden, bias, hb);
    k1_scores<<<NROWS / 32, 64, 0, stream>>>(out_t, Wb, hb, vvec, scores);
    k2_softmax<<<BATCH, 256, 0, stream>>>(scores, attn);
    k3_context<<<dim3(HDIM / 128, BATCH), 256, 0, stream>>>(out_t, attn, ctx);
}
